// CausalMultiHeadSelfAttention_14946486190805
// MI455X (gfx1250) — matrix operand order repaired and checked by reading, not compiled, not run on hardware
//
#include <hip/hip_runtime.h>
#include <hip/hip_bf16.h>
#include <cstdint>

typedef __attribute__((ext_vector_type(16))) __bf16 v16bf;
typedef __attribute__((ext_vector_type(8)))  float  v8f;

#define WMMA_BF16(A, B, C) \
  __builtin_amdgcn_wmma_f32_16x16x32_bf16(false, (A), false, (B), (short)0, (C), false, false)

// Reduce across the 16-lane half-wave that co-owns a row in the WMMA C layout.
__device__ __forceinline__ float half16_max(float v) {
#pragma unroll
  for (int m = 1; m < 16; m <<= 1) v = fmaxf(v, __shfl_xor(v, m, 32));
  return v;
}
__device__ __forceinline__ float half16_sum(float v) {
#pragma unroll
  for (int m = 1; m < 16; m <<= 1) v += __shfl_xor(v, m, 32);
  return v;
}

// ---------------------------------------------------------------------------
// Elementwise fp32 -> bf16 (one-time operand conversion, hoisted out of GEMMs)
// ---------------------------------------------------------------------------
__global__ __launch_bounds__(256) void f32_to_bf16(const float* __restrict__ in,
                                                   __bf16* __restrict__ out, int n) {
  const int i = blockIdx.x * blockDim.x + threadIdx.x;
  if (i < n) out[i] = (__bf16)in[i];
}

// ---------------------------------------------------------------------------
// C[M,N] = A[M,K] @ W[N,K]^T, bf16 operands, f32 accum/output.
// One wave computes a 16x64 strip (4 accumulators per A fetch).
// Double-buffered K loop: iteration k+1's tiles load into `nxt` while the four
// v_wmma_f32_16x16x32_bf16 of iteration k consume `cur`, so per-wave loads
// overlap the matrix pipe instead of s_wait_loadcnt 0 before every WMMA.
// ---------------------------------------------------------------------------
struct Tiles { v16bf a, b0, b1, b2, b3; };

__device__ __forceinline__ void load_tiles(Tiles& t, const __bf16* __restrict__ ap,
                                           const __bf16* __restrict__ bp,
                                           size_t bstride, int k0) {
#pragma unroll
  for (int i = 0; i < 16; ++i) {
    t.a[i]     = ap[k0 + i];
  }
#pragma unroll
  for (int i = 0; i < 16; ++i) {
    t.b0[i] = bp[k0 + i];
    t.b1[i] = bp[bstride + k0 + i];
    t.b2[i] = bp[2 * bstride + k0 + i];
    t.b3[i] = bp[3 * bstride + k0 + i];
  }
}

__global__ __launch_bounds__(256) void gemm_bf16_16x64(
    const __bf16* __restrict__ A, const __bf16* __restrict__ W, float* __restrict__ C,
    int M, int N, int K) {
  const int lane    = threadIdx.x & 31;
  const int wave    = (blockIdx.x * blockDim.x + threadIdx.x) >> 5;
  const int groupsN = N >> 6;
  const int nwaves  = (M >> 4) * groupsN;
  if (wave >= nwaves) return;                       // wave-uniform branch
  const int tm = wave / groupsN, tg = wave % groupsN;
  const int lanelo = lane & 15;
  const int hi     = lane >> 4;

  // A 16x32 layout: lanes 0-15 row=lane, elems K[0..15];
  //                 lanes 16-31 same rows, elems K[16..31].
  // B 32x16 layout: lane n=lane&15; lanes 0-15 K=0..15, lanes 16-31 K=16..31.
  const __bf16* ap = A + (size_t)(tm * 16 + lanelo) * K + (hi ? 16 : 0);
  const __bf16* bp = W + (size_t)(tg * 64 + lanelo) * K + (hi ? 16 : 0);
  const size_t bstride = (size_t)16 * K;

  v8f ac0 = {}, ac1 = {}, ac2 = {}, ac3 = {};
  Tiles cur, nxt;
  load_tiles(cur, ap, bp, bstride, 0);
  int k0 = 0;
  for (; k0 + 32 < K; k0 += 32) {
    load_tiles(nxt, ap, bp, bstride, k0 + 32);      // in flight behind WMMAs
    __builtin_prefetch(ap + k0 + 96, 0, 1);         // global_prefetch_b8
    __builtin_prefetch(bp + k0 + 96, 0, 1);
    ac0 = WMMA_BF16(cur.a, cur.b0, ac0);
    ac1 = WMMA_BF16(cur.a, cur.b1, ac1);
    ac2 = WMMA_BF16(cur.a, cur.b2, ac2);
    ac3 = WMMA_BF16(cur.a, cur.b3, ac3);
    cur = nxt;                                      // register rotation
  }
  ac0 = WMMA_BF16(cur.a, cur.b0, ac0);
  ac1 = WMMA_BF16(cur.a, cur.b1, ac1);
  ac2 = WMMA_BF16(cur.a, cur.b2, ac2);
  ac3 = WMMA_BF16(cur.a, cur.b3, ac3);

  float* cp = C + (size_t)(tm * 16 + (hi ? 8 : 0)) * N + tg * 64 + lanelo;
#pragma unroll
  for (int r = 0; r < 8; ++r) {
    cp[(size_t)r * N + 0]  = ac0[r];
    cp[(size_t)r * N + 16] = ac1[r];
    cp[(size_t)r * N + 32] = ac2[r];
    cp[(size_t)r * N + 48] = ac3[r];
  }
}

// ---------------------------------------------------------------------------
// RoPE (fp32 math) on q,k + head split to bf16.
// q,k -> [B*H, S, 64]; v -> TRANSPOSED [B*H, 64, S] so the PV B-operand in
// attention reads contiguous 32B per lane (same pattern as the K-operand).
// ---------------------------------------------------------------------------
__global__ __launch_bounds__(256) void rope_split(
    const float* __restrict__ qkv, const int* __restrict__ pos,
    __bf16* __restrict__ qh, __bf16* __restrict__ kh, __bf16* __restrict__ vT) {
  const int gid = blockIdx.x * blockDim.x + threadIdx.x;
  const int i = gid & 31;
  const int s = (gid >> 5) & 2047;
  const int h = (gid >> 16) & 15;
  const int b = gid >> 20;
  const int bh = b * 16 + h;
  const float p   = (float)pos[s];
  const float ang = p * __expf(-(float)i * (9.210340371976184f / 32.0f)); // 10000^(-i/32)
  const float cs = __cosf(ang), sn = __sinf(ang);
  const size_t src = ((size_t)(b * 2048 + s)) * 3072 + h * 64 + 2 * i;
  const size_t dst = ((size_t)bh * 2048 + s) * 64 + 2 * i;
  const float q1 = qkv[src], q2 = qkv[src + 1];
  qh[dst]     = (__bf16)(q1 * cs - q2 * sn);
  qh[dst + 1] = (__bf16)(q1 * sn + q2 * cs);
  const float k1 = qkv[src + 1024], k2 = qkv[src + 1025];
  kh[dst]     = (__bf16)(k1 * cs - k2 * sn);
  kh[dst + 1] = (__bf16)(k1 * sn + k2 * cs);
  vT[((size_t)bh * 64 + 2 * i) * 2048 + s]     = (__bf16)qkv[src + 2048];
  vT[((size_t)bh * 64 + 2 * i + 1) * 2048 + s] = (__bf16)qkv[src + 2049];
}

// ---------------------------------------------------------------------------
// Causal flash attention, one wave per (b, h, 16-row query tile).
// 32-key chunks: QK^T = 4 WMMAs, online softmax (half-wave row reductions),
// P staged through LDS (C layout -> A operand), PV = 4 WMMAs from V^T.
// ---------------------------------------------------------------------------
__global__ __launch_bounds__(256) void flash_attn(
    const __bf16* __restrict__ qh, const __bf16* __restrict__ kh,
    const __bf16* __restrict__ vT, __bf16* __restrict__ y) {
  __shared__ __bf16 pbuf[8 * 16 * 32];              // 1KB per wave
  const int lane   = threadIdx.x & 31;
  const int wslot  = threadIdx.x >> 5;
  const int wave   = (blockIdx.x * blockDim.x + threadIdx.x) >> 5;  // 0..8191
  const int qt = wave & 127;
  const int bh = wave >> 7;
  const int b  = bh >> 4, h = bh & 15;
  const int qs = qt * 16;
  const int lanelo = lane & 15;
  const int hi     = lane >> 4;
  const int rhalf  = hi ? 8 : 0;
  __bf16* pb = pbuf + wslot * (16 * 32);

  const __bf16* qptr = qh + ((size_t)bh * 2048 + qs + lanelo) * 64;
  const int qoff = hi ? 16 : 0;
  v16bf qa0, qa1;
#pragma unroll
  for (int i = 0; i < 16; ++i) {
    qa0[i]     = qptr[qoff + i];
    qa1[i]     = qptr[32 + qoff + i];
  }

  float m_r[8], l_r[8], alpha[8];
  v8f o0 = {}, o1 = {}, o2 = {}, o3 = {};
#pragma unroll
  for (int r = 0; r < 8; ++r) { m_r[r] = -3.0e38f; l_r[r] = 0.0f; }

  const float scale = 0.125f;                       // 1/sqrt(64)
  const int jmax = (qs + 15) >> 5;                  // only last chunk masked
  const __bf16* kbase = kh + (size_t)bh * 2048 * 64;
  const __bf16* vbase = vT + (size_t)bh * 64 * 2048;
  const int koff = hi ? 16 : 0;

  for (int j = 0; j <= jmax; ++j) {
    const int kb = j * 32;
    const __bf16* kp0 = kbase + (size_t)(kb + lanelo) * 64;
    const __bf16* kp1 = kp0 + 16 * 64;
    __builtin_prefetch(kp0 + 32 * 64, 0, 1);        // next chunk's K rows
    __builtin_prefetch(kp1 + 32 * 64, 0, 1);
    v16bf bk0a, bk0b, bk1a, bk1b;
#pragma unroll
    for (int i = 0; i < 16; ++i) {
      bk0a[i] = kp0[koff + i];
      bk0b[i] = kp0[32 + koff + i];
      bk1a[i] = kp1[koff + i];
      bk1b[i] = kp1[32 + koff + i];
    }
    v8f s0 = {}, s1 = {};
    s0 = WMMA_BF16(qa0, bk0a, s0);
    s0 = WMMA_BF16(qa1, bk0b, s0);
    s1 = WMMA_BF16(qa0, bk1a, s1);
    s1 = WMMA_BF16(qa1, bk1b, s1);

#pragma unroll
    for (int r = 0; r < 8; ++r) {
      float v0 = s0[r] * scale;
      float v1 = s1[r] * scale;
      if (j == jmax) {                              // causal mask, diagonal chunk
        const int row = qs + r + rhalf;
        if (kb + lanelo > row)      v0 = -3.0e38f;
        if (kb + 16 + lanelo > row) v1 = -3.0e38f;
      }
      const float cm = half16_max(fmaxf(v0, v1));
      const float mn = fmaxf(m_r[r], cm);
      alpha[r] = __expf(m_r[r] - mn);
      const float p0 = __expf(v0 - mn);
      const float p1 = __expf(v1 - mn);
      l_r[r] = l_r[r] * alpha[r] + half16_sum(p0 + p1);
      m_r[r] = mn;
      pb[(r + rhalf) * 32 + lanelo]      = (__bf16)p0;   // stage P for relayout
      pb[(r + rhalf) * 32 + 16 + lanelo] = (__bf16)p1;
    }
#pragma unroll
    for (int r = 0; r < 8; ++r) {
      o0[r] *= alpha[r]; o1[r] *= alpha[r]; o2[r] *= alpha[r]; o3[r] *= alpha[r];
    }
    __builtin_amdgcn_wave_barrier();                // order DS store->load (per-wave in-order DS)
    v16bf pa;                                       // P as 16x32 A operand
#pragma unroll
  for (int i = 0; i < 16; ++i) {
      pa[i]     = pb[lanelo * 32 + qoff + i];
    }
    __builtin_amdgcn_wave_barrier();
    v16bf bv0, bv1, bv2, bv3;
    const __bf16* vp = vbase + kb + koff;
    __builtin_prefetch(vp + 32, 0, 1);              // next chunk's V columns
#pragma unroll
    for (int i = 0; i < 16; ++i) {
      bv0[i] = vp[(size_t)(lanelo + 0)  * 2048 + i];
      bv1[i] = vp[(size_t)(lanelo + 16) * 2048 + i];
      bv2[i] = vp[(size_t)(lanelo + 32) * 2048 + i];
      bv3[i] = vp[(size_t)(lanelo + 48) * 2048 + i];
    }
    o0 = WMMA_BF16(pa, bv0, o0);
    o1 = WMMA_BF16(pa, bv1, o1);
    o2 = WMMA_BF16(pa, bv2, o2);
    o3 = WMMA_BF16(pa, bv3, o3);
  }

  __bf16* yp = y + ((size_t)b * 2048 + qs + rhalf) * 1024 + h * 64 + lanelo;
#pragma unroll
  for (int r = 0; r < 8; ++r) {
    const float inv = 1.0f / l_r[r];
    yp[(size_t)r * 1024 + 0]  = (__bf16)(o0[r] * inv);
    yp[(size_t)r * 1024 + 16] = (__bf16)(o1[r] * inv);
    yp[(size_t)r * 1024 + 32] = (__bf16)(o2[r] * inv);
    yp[(size_t)r * 1024 + 48] = (__bf16)(o3[r] * inv);
  }
}

// ---------------------------------------------------------------------------
extern "C" void kernel_launch(void* const* d_in, const int* in_sizes, int n_in,
                              void* d_out, int out_size, void* d_ws, size_t ws_size,
                              hipStream_t stream) {
  const float* x     = (const float*)d_in[0];   // [4,2048,1024]
  const float* w_qkv = (const float*)d_in[1];   // [3072,1024]
  const float* w_out = (const float*)d_in[2];   // [1024,1024]
  const int*   pos   = (const int*)d_in[3];     // [2048]
  float*       out   = (float*)d_out;           // [4,2048,1024]

  char* ws = (char*)d_ws;
  float*  qkv  = (float*)ws;                                  // 96 MB fp32 [8192,3072]
  __bf16* qh   = (__bf16*)(ws + 100663296);                   // 16 MB [64,2048,64]
  __bf16* kh   = (__bf16*)(ws + 117440512);                   // 16 MB [64,2048,64]
  __bf16* vT   = (__bf16*)(ws + 134217728);                   // 16 MB [64,64,2048]
  __bf16* ybf  = (__bf16*)(ws + 150994944);                   // 16 MB [8192,1024]
  __bf16* xbf  = (__bf16*)(ws + 167772160);                   // 16 MB [8192,1024]
  __bf16* wqbf = (__bf16*)(ws + 184549376);                   //  6 MB [3072,1024]
  __bf16* wobf = (__bf16*)(ws + 190840832);                   //  2 MB [1024,1024]

  f32_to_bf16<<<32768, 256, 0, stream>>>(x,     xbf,  8388608);
  f32_to_bf16<<<12288, 256, 0, stream>>>(w_qkv, wqbf, 3145728);
  f32_to_bf16<<<4096,  256, 0, stream>>>(w_out, wobf, 1048576);

  // 1) qkv = x @ w_qkv^T : (8192/16)*(3072/64) = 24576 waves
  gemm_bf16_16x64<<<3072, 256, 0, stream>>>(xbf, wqbf, qkv, 8192, 3072, 1024);
  // 2) RoPE + head split (+ V transpose)
  rope_split<<<16384, 256, 0, stream>>>(qkv, pos, qh, kh, vT);
  // 3) flash attention: 4*16*128 = 8192 waves
  flash_attn<<<1024, 256, 0, stream>>>(qh, kh, vT, ybf);
  // 4) out = y @ w_out^T : (8192/16)*(1024/64) = 8192 waves
  gemm_bf16_16x64<<<1024, 256, 0, stream>>>(ybf, wobf, out, 8192, 1024, 1024);
}
